// ANEE_3058016715229
// MI455X (gfx1250) — matrix operand order repaired and checked by reading, not compiled, not run on hardware
//
#include <hip/hip_runtime.h>

#define N_NODES 8192
#define FDIM 128
#define NEDGE 262144
#define HDIM 64
#define NHEADS 4
#define HD 16
#define NEG_SLOPE 0.01f

typedef __attribute__((ext_vector_type(16))) _Float16 v16h;
typedef __attribute__((ext_vector_type(8)))  _Float16 v8h;
typedef __attribute__((ext_vector_type(8)))  float    v8f;

// K index inside a 16x32 f16 A-tile for element e (of 16 per lane), half hi.
// ISA 7.12.2: lanes0-15: V0..V3 -> K0..7, V4..V7 -> K16..23; lanes16-31: +8.
// Consequence: per lane the 16 elements are two contiguous 8-half runs:
//   e 0..7  <-> K = hi*8 + 0..7
//   e 8..15 <-> K = 16 + hi*8 + 0..7
__device__ __forceinline__ v16h cat8(v8h lo, v8h hi) {
    v16h r;
#pragma unroll
    for (int i = 0; i < 8; ++i) { r[i] = lo[i]; r[8 + i] = hi[i]; }
    return r;
}

// Row reductions across the 16 lanes of a half-wave (same C/D rows).
__device__ __forceinline__ float half_max(float v) {
    v = fmaxf(v, __shfl_xor(v, 1, 32));
    v = fmaxf(v, __shfl_xor(v, 2, 32));
    v = fmaxf(v, __shfl_xor(v, 4, 32));
    v = fmaxf(v, __shfl_xor(v, 8, 32));
    return v;
}
__device__ __forceinline__ float half_sum(float v) {
    v += __shfl_xor(v, 1, 32);
    v += __shfl_xor(v, 2, 32);
    v += __shfl_xor(v, 4, 32);
    v += __shfl_xor(v, 8, 32);
    return v;
}

// ---------- staging kernels ----------
__global__ void f32_to_f16_kernel(const float* __restrict__ src, _Float16* __restrict__ dst, int n) {
    int i = blockIdx.x * blockDim.x + threadIdx.x;
    if (i < n) dst[i] = (_Float16)src[i];
}

// W [R][C] f32 -> Wt [C][R] f16
__global__ void transpose_to_f16_kernel(const float* __restrict__ W, _Float16* __restrict__ Wt,
                                        int R, int C) {
    int i = blockIdx.x * blockDim.x + threadIdx.x;
    if (i >= R * C) return;
    int r = i / C, c = i % C;
    Wt[c * R + r] = (_Float16)W[r * C + c];
}

// qkv16 [N][192] -> Qh [H][N][16] (scaled 1/sqrt(HD)), Kt [H][16][N], Vh [H][N][16]
__global__ void pack_qkv_kernel(const _Float16* __restrict__ qkv16,
                                _Float16* __restrict__ Qh, _Float16* __restrict__ Kt,
                                _Float16* __restrict__ Vh) {
    int i = blockIdx.x * blockDim.x + threadIdx.x;
    if (i >= NHEADS * N_NODES) return;
    int head = i >> 13, node = i & (N_NODES - 1);
    const _Float16* row = qkv16 + node * 192;
#pragma unroll
    for (int d = 0; d < HD; ++d) {
        Qh[(head * N_NODES + node) * HD + d] = (_Float16)(0.25f * (float)row[head * HD + d]);
        Kt[(head * HD + d) * N_NODES + node] = row[HDIM + head * HD + d];
        Vh[(head * N_NODES + node) * HD + d] = row[2 * HDIM + head * HD + d];
    }
}

// ---------- f16 GEMM: C = opt_leaky(A[MxK] @ Bt[KxN] + bias) ----------
// A, Bt pre-staged f16; one wave per 16x16 tile; A fragment loads are b128, B is 16 strided half loads.
__global__ void gemm_f16_kernel(const _Float16* __restrict__ A, const _Float16* __restrict__ Bt,
                                const float* __restrict__ bias, float* __restrict__ C32,
                                _Float16* __restrict__ C16, int M, int Ncols, int K,
                                int apply_leaky) {
    int gw = (blockIdx.x * blockDim.x + threadIdx.x) >> 5;
    int tilesM = M >> 4, tilesN = Ncols >> 4;
    if (gw >= tilesM * tilesN) return;
    int tm = gw % tilesM, tn = gw / tilesM;
    int row0 = tm << 4, col0 = tn << 4;
    int lane = threadIdx.x & 31;
    int m = lane & 15, hi = lane >> 4;

    v8f acc;
    float b = bias ? bias[col0 + m] : 0.0f;
#pragma unroll
    for (int r = 0; r < 8; ++r) acc[r] = b;

    for (int k0 = 0; k0 < K; k0 += 32) {
        const _Float16* arow = A + (row0 + m) * K + k0;
        v8h alo = *(const v8h*)(arow + hi * 8);
        v8h ahi = *(const v8h*)(arow + 16 + hi * 8);
        const _Float16* brow = Bt + k0 * Ncols + col0 + m; // B[k][n=m], k strided by Ncols
        v16h bfr;
#pragma unroll
        for (int i = 0; i < 8; ++i) {
            bfr[i] = brow[(hi * 8 + i) * Ncols];
            bfr[8 + i] = brow[(16 + hi * 8 + i) * Ncols];
        }
        acc = __builtin_amdgcn_wmma_f32_16x16x32_f16(false, cat8(alo, ahi), false, bfr,
                                                     (short)0, acc, false, false);
    }
#pragma unroll
    for (int r = 0; r < 8; ++r) {
        float x = acc[r];
        if (apply_leaky) x = (x >= 0.0f) ? x : NEG_SLOPE * x;
        int idx = (row0 + hi * 8 + r) * Ncols + col0 + m;
        if (C32) C32[idx] = x;
        if (C16) C16[idx] = (_Float16)x;
    }
}

// ---------- flash attention, 32 keys / iteration ----------
__global__ void attn_flash_kernel(const _Float16* __restrict__ Qh, const _Float16* __restrict__ Kt,
                                  const _Float16* __restrict__ Vh, _Float16* __restrict__ o16) {
    __shared__ __align__(16) _Float16 pbuf[8][16][32];
    int gw = (blockIdx.x * blockDim.x + threadIdx.x) >> 5;
    int w = (threadIdx.x >> 5) & 7;
    int lane = threadIdx.x & 31;
    int m = lane & 15, hi = lane >> 4;
    const int qtiles = N_NODES / 16;
    if (gw >= NHEADS * qtiles) return;
    int head = gw / qtiles;
    int row0 = (gw % qtiles) << 4;

    // Q fragment: K=0..15 live (one b128 per lane), K=16..31 zero padding
    v16h aq;
    {
        v8h qlo = *(const v8h*)(Qh + (head * N_NODES + row0 + m) * HD + hi * 8);
#pragma unroll
        for (int i = 0; i < 8; ++i) { aq[i] = qlo[i]; aq[8 + i] = (_Float16)0.0f; }
    }

    float mrow[8], lrow[8];
    v8f accO;
#pragma unroll
    for (int r = 0; r < 8; ++r) { mrow[r] = -1e30f; lrow[r] = 0.0f; accO[r] = 0.0f; }

    const _Float16* kth = Kt + head * HD * N_NODES;
    const _Float16* vh  = Vh + head * N_NODES * HD;

    for (int key0 = 0; key0 < N_NODES; key0 += 32) {
        if (key0 + 32 < N_NODES) {
            __builtin_prefetch(kth + lane * N_NODES + key0 + 32, 0, 3);
            __builtin_prefetch(vh + (key0 + 32 + lane) * HD, 0, 3);
        }
        // two 16-key score tiles: B[d=hi*8+e][n=m] = Kt[head][d][key0+16t+m]
        v8f s0, s1;
#pragma unroll
        for (int r = 0; r < 8; ++r) { s0[r] = 0.0f; s1[r] = 0.0f; }
        {
            v16h bk0, bk1;
#pragma unroll
            for (int i = 0; i < 8; ++i) {
                const _Float16* kp = kth + (hi * 8 + i) * N_NODES + key0 + m;
                bk0[i] = kp[0];
                bk1[i] = kp[16];
                bk0[8 + i] = (_Float16)0.0f;
                bk1[8 + i] = (_Float16)0.0f;
            }
            s0 = __builtin_amdgcn_wmma_f32_16x16x32_f16(false, aq, false, bk0, (short)0, s0, false, false);
            s1 = __builtin_amdgcn_wmma_f32_16x16x32_f16(false, aq, false, bk1, (short)0, s1, false, false);
        }
        // online softmax over the 32 key columns
#pragma unroll
        for (int r = 0; r < 8; ++r) {
            float rmax = fmaxf(half_max(s0[r]), half_max(s1[r]));
            float mnew = fmaxf(mrow[r], rmax);
            float p0 = __expf(s0[r] - mnew);
            float p1 = __expf(s1[r] - mnew);
            float rsum = half_sum(p0 + p1);
            float csc = __expf(mrow[r] - mnew);
            lrow[r] = lrow[r] * csc + rsum;
            accO[r] *= csc;
            mrow[r] = mnew;
            pbuf[w][hi * 8 + r][m] = (_Float16)p0;
            pbuf[w][hi * 8 + r][16 + m] = (_Float16)p1;
        }
        asm volatile("s_wait_dscnt 0x0" ::: "memory");
        // P fragment from LDS: two contiguous 8-half runs per lane (ds_load_b128)
        v16h ap = cat8(*(const v8h*)&pbuf[w][m][hi * 8],
                       *(const v8h*)&pbuf[w][m][16 + hi * 8]);
        // V fragment: full K=32, B[k][n=m] = Vh[head][key0+k][m], k in table order
        const _Float16* vp = vh + key0 * HD + m;
        v16h bv;
#pragma unroll
        for (int i = 0; i < 8; ++i) {
            bv[i] = vp[(hi * 8 + i) * HD];
            bv[8 + i] = vp[(16 + hi * 8 + i) * HD];
        }
        accO = __builtin_amdgcn_wmma_f32_16x16x32_f16(false, ap, false, bv, (short)0, accO, false, false);
        asm volatile("s_wait_dscnt 0x0" ::: "memory");
    }
#pragma unroll
    for (int r = 0; r < 8; ++r)
        o16[(row0 + hi * 8 + r) * HDIM + head * HD + m] = (_Float16)(accO[r] / lrow[r]);
}

// ---------- edge stage ----------
__global__ void edge_ei_kernel(const float* __restrict__ h_bar, const int* __restrict__ edges,
                               const float* __restrict__ edge_features,
                               const float* __restrict__ attn_vec, const float* __restrict__ W_e,
                               float* __restrict__ e_i) {
    int e = blockIdx.x * blockDim.x + threadIdx.x;
    if (e >= NEDGE) return;
    int src = edges[2 * e], tgt = edges[2 * e + 1];
    float scal = 0.0f;
#pragma unroll 8
    for (int j = 0; j < HDIM; ++j)
        scal += attn_vec[j] * h_bar[src * HDIM + j] + attn_vec[HDIM + j] * h_bar[tgt * HDIM + j];
    float ef = edge_features[e];
#pragma unroll 8
    for (int j = 0; j < HDIM; ++j) {
        float z = ef * W_e[j] * scal;
        e_i[e * HDIM + j] = 1.0f / (1.0f + __expf(-z));
    }
}

// w = softmax(e_i @ W_m, axis=1); contrib = w * h_bar[tgt]; scatter-add by src.
__global__ void edge_msg_kernel(const float* __restrict__ e_i, const _Float16* __restrict__ Wm16,
                                const float* __restrict__ h_bar, const int* __restrict__ edges,
                                float* __restrict__ h_u) {
    int gw = (blockIdx.x * blockDim.x + threadIdx.x) >> 5;
    if (gw >= NEDGE / 16) return;
    int e0 = gw << 4;
    int lane = threadIdx.x & 31;
    int m = lane & 15, hi = lane >> 4;

    v8f acc[4];
#pragma unroll
    for (int ct = 0; ct < 4; ++ct)
#pragma unroll
        for (int r = 0; r < 8; ++r) acc[ct][r] = 0.0f;

    for (int k0 = 0; k0 < HDIM; k0 += 32) {
        // A fragment: e_i rows are f32 (it is a required f32 output) -> convert on load
        v16h av;
        const float* arow = e_i + (e0 + m) * HDIM + k0;
#pragma unroll
        for (int i = 0; i < 8; ++i) {
            av[i] = (_Float16)arow[hi * 8 + i];
            av[8 + i] = (_Float16)arow[16 + hi * 8 + i];
        }
#pragma unroll
        for (int ct = 0; ct < 4; ++ct) {
            const _Float16* brow = Wm16 + k0 * HDIM + ct * 16 + m; // W_m not transposed
            v16h bv;
#pragma unroll
            for (int i = 0; i < 8; ++i) {
                bv[i] = brow[(hi * 8 + i) * HDIM];
                bv[8 + i] = brow[(16 + hi * 8 + i) * HDIM];
            }
            acc[ct] = __builtin_amdgcn_wmma_f32_16x16x32_f16(false, av, false, bv,
                                                             (short)0, acc[ct], false, false);
        }
    }
    float rmax[8], rsum[8];
#pragma unroll
    for (int r = 0; r < 8; ++r) {
        float v = fmaxf(fmaxf(acc[0][r], acc[1][r]), fmaxf(acc[2][r], acc[3][r]));
        rmax[r] = half_max(v);
    }
    float pex[4][8];
#pragma unroll
    for (int r = 0; r < 8; ++r) {
        float sum = 0.0f;
#pragma unroll
        for (int ct = 0; ct < 4; ++ct) { pex[ct][r] = __expf(acc[ct][r] - rmax[r]); sum += pex[ct][r]; }
        rsum[r] = half_sum(sum);
    }
    int srcv[8], tgtv[8];
#pragma unroll
    for (int r = 0; r < 8; ++r) {
        int ee = e0 + hi * 8 + r;
        srcv[r] = edges[2 * ee];
        tgtv[r] = edges[2 * ee + 1];
    }
#pragma unroll
    for (int ct = 0; ct < 4; ++ct) {
        int col = ct * 16 + m;
#pragma unroll
        for (int r = 0; r < 8; ++r) {
            float wv = pex[ct][r] / rsum[r];
            atomicAdd(&h_u[srcv[r] * HDIM + col], wv * h_bar[tgtv[r] * HDIM + col]);
        }
    }
}

__global__ void zero_kernel(float* __restrict__ p, int n) {
    int i = blockIdx.x * blockDim.x + threadIdx.x;
    if (i < n) p[i] = 0.0f;
}

__global__ void leaky_out_kernel(const float* __restrict__ h_u, float* __restrict__ out, int n) {
    int i = blockIdx.x * blockDim.x + threadIdx.x;
    if (i < n) {
        float x = h_u[i];
        out[i] = (x >= 0.0f) ? x : NEG_SLOPE * x;
    }
}

extern "C" void kernel_launch(void* const* d_in, const int* in_sizes, int n_in,
                              void* d_out, int out_size, void* d_ws, size_t ws_size,
                              hipStream_t stream) {
    const float* node_features = (const float*)d_in[0];
    const float* edge_features = (const float*)d_in[1];
    const int*   edges         = (const int*)d_in[2];
    const float* W_u           = (const float*)d_in[3];
    const float* in_proj_w     = (const float*)d_in[4];
    const float* in_proj_b     = (const float*)d_in[5];
    const float* out_proj_w    = (const float*)d_in[6];
    const float* out_proj_b    = (const float*)d_in[7];
    const float* W_e           = (const float*)d_in[8];
    const float* attn_vec      = (const float*)d_in[9];
    const float* W_m           = (const float*)d_in[10];

    // ---- workspace layout (halves first, then floats; ~15 MB total) ----
    _Float16* f16ws = (_Float16*)d_ws;
    _Float16* Xh     = f16ws;                      // 8192*128  = 1048576
    _Float16* h16    = Xh + N_NODES * FDIM;        // 8192*64   =  524288
    _Float16* qkv16  = h16 + N_NODES * HDIM;       // 8192*192  = 1572864
    _Float16* Qh     = qkv16 + N_NODES * 3 * HDIM; // 4*8192*16 =  524288
    _Float16* Kt     = Qh + NHEADS * N_NODES * HD; //            524288
    _Float16* Vh     = Kt + NHEADS * N_NODES * HD; //            524288
    _Float16* o16    = Vh + NHEADS * N_NODES * HD; //            524288
    _Float16* Wt_u   = o16 + N_NODES * HDIM;       // 128*64
    _Float16* Wt_in  = Wt_u + FDIM * HDIM;         // 64*192
    _Float16* Wt_out = Wt_in + HDIM * 3 * HDIM;    // 64*64
    _Float16* Wm16   = Wt_out + HDIM * HDIM;       // 64*64
    float*    f32ws  = (float*)(Wm16 + HDIM * HDIM);
    float*    h_bar  = f32ws;                      // 8192*64
    float*    h_u    = h_bar + N_NODES * HDIM;     // 8192*64

    float* out_hu = (float*)d_out;
    float* out_ei = out_hu + N_NODES * HDIM;

    dim3 blk(256);

    // 0) stage f16 operands
    f32_to_f16_kernel<<<dim3(N_NODES * FDIM / 256), blk, 0, stream>>>(node_features, Xh, N_NODES * FDIM);
    transpose_to_f16_kernel<<<dim3(HDIM * FDIM / 256), blk, 0, stream>>>(W_u, Wt_u, HDIM, FDIM);
    transpose_to_f16_kernel<<<dim3(3 * HDIM * HDIM / 256), blk, 0, stream>>>(in_proj_w, Wt_in, 3 * HDIM, HDIM);
    transpose_to_f16_kernel<<<dim3(HDIM * HDIM / 256), blk, 0, stream>>>(out_proj_w, Wt_out, HDIM, HDIM);
    f32_to_f16_kernel<<<dim3(HDIM * HDIM / 256), blk, 0, stream>>>(W_m, Wm16, HDIM * HDIM);

    // 1) h = leaky(X @ W_u^T)  (f16 out only; consumed by next GEMM)
    { int waves = (N_NODES / 16) * (HDIM / 16);
      gemm_f16_kernel<<<dim3(waves * 32 / 256), blk, 0, stream>>>(
          Xh, Wt_u, nullptr, nullptr, h16, N_NODES, HDIM, FDIM, 1); }
    // 2) qkv = h @ in_proj_w^T + b  (f16 out; consumed by pack)
    { int waves = (N_NODES / 16) * (3 * HDIM / 16);
      gemm_f16_kernel<<<dim3(waves * 32 / 256), blk, 0, stream>>>(
          h16, Wt_in, in_proj_b, nullptr, qkv16, N_NODES, 3 * HDIM, HDIM, 0); }
    // 3) pack per-head Q(scaled)/K^T/V
    pack_qkv_kernel<<<dim3(NHEADS * N_NODES / 256), blk, 0, stream>>>(qkv16, Qh, Kt, Vh);
    // 4) flash attention -> o16
    { int waves = NHEADS * (N_NODES / 16);
      attn_flash_kernel<<<dim3(waves * 32 / 256), blk, 0, stream>>>(Qh, Kt, Vh, o16); }
    // 5) h_bar = o @ out_proj_w^T + b  (f32 out for edge stage)
    { int waves = (N_NODES / 16) * (HDIM / 16);
      gemm_f16_kernel<<<dim3(waves * 32 / 256), blk, 0, stream>>>(
          o16, Wt_out, out_proj_b, h_bar, nullptr, N_NODES, HDIM, HDIM, 0); }
    // 6) e_i (second output, also consumed by stage 8)
    edge_ei_kernel<<<dim3(NEDGE / 256), blk, 0, stream>>>(
        h_bar, edges, edge_features, attn_vec, W_e, out_ei);
    // 7) zero segment-sum accumulator
    zero_kernel<<<dim3(N_NODES * HDIM / 256), blk, 0, stream>>>(h_u, N_NODES * HDIM);
    // 8) edge softmax GEMM + scatter-add
    { int waves = NEDGE / 16;
      edge_msg_kernel<<<dim3(waves * 32 / 256), blk, 0, stream>>>(
          out_ei, Wm16, h_bar, edges, h_u); }
    // 9) first output = leaky(h_u)
    leaky_out_kernel<<<dim3(N_NODES * HDIM / 256), blk, 0, stream>>>(h_u, out_hu, N_NODES * HDIM);
}
